// QuantizedLinear_41394894798994
// MI455X (gfx1250) — 8-bit matrix operand order repaired and checked by reading, compiled, not run on hardware
//
#include <hip/hip_runtime.h>
#include <stdint.h>

typedef __attribute__((ext_vector_type(8))) int v8i;
typedef __attribute__((ext_vector_type(4))) int v4i;
typedef __attribute__((ext_vector_type(2))) int v2i;
typedef int   gvi4 __attribute__((vector_size(16)));
typedef float gvf4 __attribute__((vector_size(16)));

#define M_TOT 8192
#define N_TOT 4096
#define K_TOT 4096

#define BLK_M 256
#define BLK_N 128
#define BLK_K 64
#define LDS_STRIDE 20

#if defined(__gfx1250__) && __has_builtin(__builtin_amdgcn_global_load_async_to_lds_b128)
#define HAVE_ASYNC 1
#else
#define HAVE_ASYNC 0
#endif

#if HAVE_ASYNC
__device__ __forceinline__ void cp_async16(const char* g, int* l) {
    __builtin_amdgcn_global_load_async_to_lds_b128(
        (__attribute__((address_space(1))) gvi4*)g,
        (__attribute__((address_space(3))) gvi4*)l,   0,   0);
}
__device__ __forceinline__ void wait_async0() {
#if __has_builtin(__builtin_amdgcn_s_wait_asynccnt)
    __builtin_amdgcn_s_wait_asynccnt(0);
#else
    asm volatile("s_wait_asynccnt 0" ::: "memory");
#endif
}
#endif

__global__ void init_kernel(unsigned* xabs) {
    if (threadIdx.x == 0 && blockIdx.x == 0) *xabs = 0u;
}

__global__ void absmax_x_kernel(const float4* __restrict__ x, unsigned* __restrict__ xabs, int n4) {
    float m = 0.f;
    for (int i = blockIdx.x * blockDim.x + threadIdx.x; i < n4; i += gridDim.x * blockDim.x) {
        float4 v = x[i];
        m = fmaxf(m, fmaxf(fmaxf(fabsf(v.x), fabsf(v.y)), fmaxf(fabsf(v.z), fabsf(v.w))));
    }
    __shared__ float red[256];
    red[threadIdx.x] = m;
    __syncthreads();
    for (int s = 128; s > 0; s >>= 1) {
        if (threadIdx.x < s) red[threadIdx.x] = fmaxf(red[threadIdx.x], red[threadIdx.x + s]);
        __syncthreads();
    }
    if (threadIdx.x == 0) atomicMax(xabs, __float_as_uint(red[0]));
}

__global__ void quant_w_kernel(const float* __restrict__ w, char* __restrict__ wq,
                               float* __restrict__ wscale) {
    const int n = blockIdx.x;
    const float4* row4 = (const float4*)(w + (size_t)n * K_TOT);
    float m = 0.f;
    for (int k = threadIdx.x; k < K_TOT / 4; k += 256) {
        float4 v = row4[k];
        m = fmaxf(m, fmaxf(fmaxf(fabsf(v.x), fabsf(v.y)), fmaxf(fabsf(v.z), fabsf(v.w))));
    }
    __shared__ float red[256];
    red[threadIdx.x] = m;
    __syncthreads();
    for (int s = 128; s > 0; s >>= 1) {
        if (threadIdx.x < s) red[threadIdx.x] = fmaxf(red[threadIdx.x], red[threadIdx.x + s]);
        __syncthreads();
    }
    __shared__ float s_scale;
    if (threadIdx.x == 0) {
        float sc = fmaxf(red[0] / 127.f, 1e-5f);
        wscale[n] = sc;
        s_scale = sc;
    }
    __syncthreads();
    const float inv = 1.f / s_scale;
    char4* qrow = (char4*)(wq + (size_t)n * K_TOT);
    for (int k = threadIdx.x; k < K_TOT / 4; k += 256) {
        float4 v = row4[k];
        int a = (int)fminf(fmaxf(rintf(v.x * inv), -128.f), 127.f);
        int b = (int)fminf(fmaxf(rintf(v.y * inv), -128.f), 127.f);
        int c = (int)fminf(fmaxf(rintf(v.z * inv), -128.f), 127.f);
        int d = (int)fminf(fmaxf(rintf(v.w * inv), -128.f), 127.f);
        qrow[k] = make_char4((char)a, (char)b, (char)c, (char)d);
    }
}

__global__ void quant_x_kernel(const gvf4* __restrict__ x, const unsigned* __restrict__ xabs,
                               char4* __restrict__ xq, int n4) {
    const float inv = 127.f / __uint_as_float(*xabs);
    for (int i = blockIdx.x * blockDim.x + threadIdx.x; i < n4; i += gridDim.x * blockDim.x) {
        gvf4 v = __builtin_nontemporal_load((gvf4*)&x[i]);
        int a = (int)fminf(fmaxf(rintf(v[0] * inv), -128.f), 127.f);
        int b = (int)fminf(fmaxf(rintf(v[1] * inv), -128.f), 127.f);
        int c = (int)fminf(fmaxf(rintf(v[2] * inv), -128.f), 127.f);
        int d = (int)fminf(fmaxf(rintf(v[3] * inv), -128.f), 127.f);
        xq[i] = make_char4((char)a, (char)b, (char)c, (char)d);
    }
}

__global__ __launch_bounds__(256) void gemm_i8_wmma_kernel(
    const char* __restrict__ Xq, const char* __restrict__ Wq,
    const float* __restrict__ wscale, const unsigned* __restrict__ xabs,
    const float* __restrict__ bias, float* __restrict__ out) {

    __shared__ __align__(16) int As[2][BLK_M * LDS_STRIDE];
    __shared__ __align__(16) int Bs[2][BLK_N * LDS_STRIDE];

    const int tid  = threadIdx.x;
    const int lane = tid & 31;
    const int wave = tid >> 5;
    const int waveM = (wave >> 1) * 64;
    const int waveN = (wave & 1) * 64;
    const int tileM = blockIdx.y * BLK_M;
    const int tileN = blockIdx.x * BLK_N;
    const int r  = lane & 15;
    const int hi = lane >> 4;

    const int ldRow = tid >> 2;
    const int ldCol = tid & 3;
    const char* aG = Xq + (size_t)(tileM + ldRow) * K_TOT + ldCol * 16;
    const char* bG = Wq + (size_t)(tileN + ldRow) * K_TOT + ldCol * 16;

    v8i acc[4][4] = {};

#if HAVE_ASYNC
    #pragma unroll
    for (int i = 0; i < 4; ++i)
        cp_async16(aG + (size_t)i * 64 * K_TOT, &As[0][(ldRow + 64 * i) * LDS_STRIDE + ldCol * 4]);
    #pragma unroll
    for (int i = 0; i < 2; ++i)
        cp_async16(bG + (size_t)i * 64 * K_TOT, &Bs[0][(ldRow + 64 * i) * LDS_STRIDE + ldCol * 4]);
#else
    #pragma unroll
    for (int i = 0; i < 4; ++i)
        *(int4*)&As[0][(ldRow + 64 * i) * LDS_STRIDE + ldCol * 4] =
            *(const int4*)(aG + (size_t)i * 64 * K_TOT);
    #pragma unroll
    for (int i = 0; i < 2; ++i)
        *(int4*)&Bs[0][(ldRow + 64 * i) * LDS_STRIDE + ldCol * 4] =
            *(const int4*)(bG + (size_t)i * 64 * K_TOT);
#endif

    const int KT = K_TOT / BLK_K;
    #pragma unroll 1
    for (int kt = 0; kt < KT; ++kt) {
        const int buf = kt & 1;

#if HAVE_ASYNC
        wait_async0();
        __syncthreads();
        if (kt + 1 < KT) {
            const int nb = buf ^ 1;
            const size_t k0 = (size_t)(kt + 1) * BLK_K;
            #pragma unroll
            for (int i = 0; i < 4; ++i)
                cp_async16(aG + (size_t)i * 64 * K_TOT + k0,
                           &As[nb][(ldRow + 64 * i) * LDS_STRIDE + ldCol * 4]);
            #pragma unroll
            for (int i = 0; i < 2; ++i)
                cp_async16(bG + (size_t)i * 64 * K_TOT + k0,
                           &Bs[nb][(ldRow + 64 * i) * LDS_STRIDE + ldCol * 4]);
        }
#else
        __syncthreads();
        int4 pa[4], pb[2];
        const bool pf = (kt + 1 < KT);
        if (pf) {
            const size_t k0 = (size_t)(kt + 1) * BLK_K;
            #pragma unroll
            for (int i = 0; i < 4; ++i) pa[i] = *(const int4*)(aG + (size_t)i * 64 * K_TOT + k0);
            #pragma unroll
            for (int i = 0; i < 2; ++i) pb[i] = *(const int4*)(bG + (size_t)i * 64 * K_TOT + k0);
        }
#endif

        v8i afrag[4];
        #pragma unroll
        for (int mt = 0; mt < 4; ++mt) {
            const int* p = &As[buf][(waveM + mt * 16 + r) * LDS_STRIDE + 2 * hi];
            v8i f;
            #pragma unroll
            for (int j = 0; j < 4; ++j) {
                v2i t = *(const v2i*)(p + 4 * j);
                f[2 * j] = t.x; f[2 * j + 1] = t.y;
            }
            afrag[mt] = f;
        }

        #pragma unroll
        for (int nt = 0; nt < 4; ++nt) {
            const int* p = &Bs[buf][(waveN + nt * 16 + r) * LDS_STRIDE + 2 * hi];
            v8i bfrag;
            #pragma unroll
            for (int j = 0; j < 4; ++j) {
                v2i t = *(const v2i*)(p + 4 * j);
                bfrag[2 * j] = t.x; bfrag[2 * j + 1] = t.y;
            }
            #pragma unroll
            for (int mt = 0; mt < 4; ++mt)
                acc[mt][nt] = __builtin_amdgcn_wmma_i32_16x16x64_iu8(
                    true, afrag[mt], true, bfrag, acc[mt][nt], false, false);
        }

#if !HAVE_ASYNC
        if (pf) {
            const int nb = buf ^ 1;
            #pragma unroll
            for (int i = 0; i < 4; ++i)
                *(int4*)&As[nb][(ldRow + 64 * i) * LDS_STRIDE + ldCol * 4] = pa[i];
            #pragma unroll
            for (int i = 0; i < 2; ++i)
                *(int4*)&Bs[nb][(ldRow + 64 * i) * LDS_STRIDE + ldCol * 4] = pb[i];
        }
#endif
    }

    const float xs = __uint_as_float(*xabs) * (1.f / 127.f);
    #pragma unroll
    for (int nt = 0; nt < 4; ++nt) {
        const int n = tileN + waveN + nt * 16 + r;
        const float sc = xs * wscale[n];
        const float bv = bias[n];
        #pragma unroll
        for (int mt = 0; mt < 4; ++mt) {
            const int mBase = tileM + waveM + mt * 16 + hi * 8;
            #pragma unroll
            for (int i = 0; i < 8; ++i)
                __builtin_nontemporal_store((float)acc[mt][nt][i] * sc + bv,
                                            &out[(size_t)(mBase + i) * N_TOT + n]);
        }
    }
}

extern "C" void kernel_launch(void* const* d_in, const int* in_sizes, int n_in,
                              void* d_out, int out_size, void* d_ws, size_t ws_size,
                              hipStream_t stream) {
    const float* x    = (const float*)d_in[0];
    const float* w    = (const float*)d_in[1];
    const float* bias = (const float*)d_in[2];
    float* out = (float*)d_out;

    char* ws = (char*)d_ws;
    char*  Xq     = ws;
    char*  Wq     = ws + (size_t)M_TOT * K_TOT;
    float* wscale = (float*)(Wq + (size_t)N_TOT * K_TOT);
    unsigned* xabs = (unsigned*)(wscale + N_TOT);

    init_kernel<<<1, 64, 0, stream>>>(xabs);
    absmax_x_kernel<<<1024, 256, 0, stream>>>((const float4*)x, xabs, M_TOT * K_TOT / 4);
    quant_w_kernel<<<N_TOT, 256, 0, stream>>>(w, Wq, wscale);
    quant_x_kernel<<<2048, 256, 0, stream>>>((const gvf4*)x, xabs, (char4*)Xq, M_TOT * K_TOT / 4);

    dim3 grid(N_TOT / BLK_N, M_TOT / BLK_M);
    gemm_i8_wmma_kernel<<<grid, 256, 0, stream>>>(Xq, Wq, wscale, xabs, bias, out);
}
